// EDAIC_LSTM_7876970021558
// MI455X (gfx1250) — compile-verified, matrix operand order checked by reading, not run on hardware
//
#include <hip/hip_runtime.h>
#include <hip/hip_bf16.h>

typedef __attribute__((ext_vector_type(16))) _Float16 v16h;
typedef __attribute__((ext_vector_type(8)))  float    v8f;

#define DEV static __device__ __forceinline__

// ---------------------------------------------------------------- helpers
DEV float siluf(float x)   { return x / (1.f + __expf(-x)); }
DEV float sigmoidf_(float x){ return 1.f / (1.f + __expf(-x)); }
DEV float logsigf(float x) { return fminf(x, 0.f) - log1pf(__expf(-fabsf(x))); }
DEV float geluf(float x)   { return 0.5f * x * (1.f + erff(x * 0.70710678118f)); }

DEV float rmax16(float v) {
  #pragma unroll
  for (int o = 1; o < 16; o <<= 1) v = fmaxf(v, __shfl_xor(v, o, 32));
  return v;
}
DEV float rsum16(float v) {
  #pragma unroll
  for (int o = 1; o < 16; o <<= 1) v += __shfl_xor(v, o, 32);
  return v;
}

// CDNA5 WMMA f16 A-fragment (16x32) / B-fragment (32x16 given B^T row-major).
// Per ISA 7.12.2: lanes 0-15 -> M=0..15 (K 0..7,16..23 pattern), lanes 16-31
// same M with K offset +8. VGPR v holds K pair kb,kb+1 with
// kb = (v>=4?16:0) + half*8 + (v&3)*2.
DEV v16h load_frag_f32(const float* __restrict__ base, int ld, int lane) {
  int m = lane & 15, half = lane >> 4;
  v16h out;
  #pragma unroll
  for (int v = 0; v < 8; ++v) {
    int kb = ((v & 4) << 2) + half * 8 + ((v & 3) << 1);
    float2 d = *(const float2*)(base + (size_t)m * ld + kb);
    out[2 * v]     = (_Float16)d.x;
    out[2 * v + 1] = (_Float16)d.y;
  }
  return out;
}
DEV v16h load_frag_f16(const _Float16* __restrict__ base, int ld, int lane) {
  int m = lane & 15, half = lane >> 4;
  v16h out;
  #pragma unroll
  for (int v = 0; v < 8; ++v) {
    int kb = ((v & 4) << 2) + half * 8 + ((v & 3) << 1);
    out[2 * v]     = base[m * ld + kb];
    out[2 * v + 1] = base[m * ld + kb + 1];
  }
  return out;
}
DEV v8f wmma_f16(v16h a, v16h b, v8f c) {
  return __builtin_amdgcn_wmma_f32_16x16x32_f16(false, a, false, b,
                                                (short)0, c, false, false);
}

// ---------------------------------------------------------------- kernels

// h = x @ W_in.T + b_in   (x: rows x 75, W: 128 x 75)
__global__ void k_embed(const float* __restrict__ x, const float* __restrict__ Wi,
                        const float* __restrict__ bi, float* __restrict__ h) {
  __shared__ float sx[75];
  int row = blockIdx.x, tid = threadIdx.x;
  if (tid < 75) sx[tid] = x[(size_t)row * 75 + tid];
  __syncthreads();
  float acc = bi[tid];
  const float* wr = Wi + (size_t)tid * 75;
  #pragma unroll 5
  for (int j = 0; j < 75; ++j) acc += sx[j] * wr[j];
  h[(size_t)row * 128 + tid] = acc;
}

// LayerNorm over D=128, one row per block of 128 threads
__global__ void k_ln128(const float* __restrict__ in, const float* __restrict__ w,
                        float* __restrict__ out) {
  __shared__ float red[128];
  int row = blockIdx.x, tid = threadIdx.x;
  float xv = in[(size_t)row * 128 + tid];
  red[tid] = xv; __syncthreads();
  for (int s = 64; s > 0; s >>= 1) { if (tid < s) red[tid] += red[tid + s]; __syncthreads(); }
  float mu = red[0] * (1.f / 128.f); __syncthreads();
  float d = xv - mu;
  red[tid] = d * d; __syncthreads();
  for (int s = 64; s > 0; s >>= 1) { if (tid < s) red[tid] += red[tid + s]; __syncthreads(); }
  float var = red[0] * (1.f / 128.f);
  out[(size_t)row * 128 + tid] = d * rsqrtf(var + 1e-5f) * w[tid];
}

// Generic WMMA GEMM: out[M,N] = A[M,K] @ W[N,K]^T + bias (+ res).
// One wave per 16x16 tile; residual presence is a compile-time flag so the
// epilogue is branch-free.
template <bool HAS_RES>
__global__ void __launch_bounds__(32)
k_gemm16(const float* __restrict__ A, int lda, const float* __restrict__ W,
         const float* __restrict__ bias, const float* __restrict__ res,
         float* __restrict__ out, int N, int K) {
  int lane = threadIdx.x;
  int m0 = blockIdx.x * 16, n0 = blockIdx.y * 16;
  const float* abase = A + (size_t)m0 * lda;
  const float* wbase = W + (size_t)n0 * K;
  v8f c = {};
  for (int k0 = 0; k0 < K; k0 += 32) {
    v16h a = load_frag_f32(abase + k0, lda, lane);
    v16h b = load_frag_f32(wbase + k0, K, lane);
    c = wmma_f16(a, b, c);
  }
  int n = lane & 15, half = lane >> 4;
  float bv = bias[n0 + n];
  #pragma unroll
  for (int r = 0; r < 8; ++r) {
    size_t idx = (size_t)(m0 + r + 8 * half) * N + n0 + n;
    float v = c[r] + bv;
    if (HAS_RES) v += res[idx];
    out[idx] = v;
  }
}

// silu(causal_conv(x, k(C,4), b)), per (row, channel)
__global__ void k_conv(const float* __restrict__ in, int ofs, int ldin,
                       const float* __restrict__ kern, const float* __restrict__ cb,
                       float* __restrict__ out, int C) {
  int i = blockIdx.x * blockDim.x + threadIdx.x;
  if (i >= 8192 * C) return;
  int row = i / C, c = i - row * C;
  int s = row & 1023;
  float acc = cb[c];
  #pragma unroll
  for (int j = 0; j < 4; ++j) {
    int sp = s - 3 + j;
    if (sp >= 0) acc += in[(size_t)(row - 3 + j) * ldin + ofs + c] * kern[c * 4 + j];
  }
  out[(size_t)row * C + c] = siluf(acc);
}

// headwise projection: out[row, h*o+oo] = dot(in[row, h*ni..], W[h,oo,:]) + bias
__global__ void k_headwise(const float* __restrict__ in, int ofs, int ldin,
                           const float* __restrict__ W, const float* __restrict__ bias,
                           float* __restrict__ out, int nh, int o, int ni) {
  int Cout = nh * o;
  int i = blockIdx.x * blockDim.x + threadIdx.x;
  if (i >= 8192 * Cout) return;
  int row = i / Cout, j = i - row * Cout;
  int hh = j / o, oo = j - hh * o;
  float acc = bias ? bias[j] : 0.f;
  const float* xp = in + (size_t)row * ldin + ofs + hh * ni;
  const float* wp = W + (size_t)(hh * o + oo) * ni;
  for (int t = 0; t < ni; ++t) acc += xp[t] * wp[t];
  out[(size_t)row * Cout + j] = acc;
}

// ig/fg gate projections over qkv concat; writes (B,4,S) layout
__global__ void k_gates(const float* __restrict__ q, const float* __restrict__ k,
                        const float* __restrict__ v, const float* __restrict__ Wig,
                        const float* __restrict__ big, const float* __restrict__ Wfg,
                        const float* __restrict__ bfg, float* __restrict__ igb,
                        float* __restrict__ fgb) {
  int i = blockIdx.x * blockDim.x + threadIdx.x;
  if (i >= 8192 * 8) return;
  int row = i >> 3, g = (i >> 1) & 3, wch = i & 1;
  const float* W = wch ? Wfg : Wig;
  float acc = wch ? bfg[g] : big[g];
  const float* wr = W + g * 768;
  const float* qr = q + (size_t)row * 256;
  const float* kr = k + (size_t)row * 256;
  const float* vr = v + (size_t)row * 256;
  for (int t = 0; t < 256; ++t)
    acc += qr[t] * wr[t] + kr[t] * wr[256 + t] + vr[t] * wr[512 + t];
  int b = row >> 10, s = row & 1023;
  (wch ? fgb : igb)[(b * 4 + g) * 1024 + s] = acc;
}

// in-place: fg -> cumsum(log_sigmoid(fg)) per (b,h); 32 serial scans
__global__ void k_fgcum(float* __restrict__ fg) {
  int bh = threadIdx.x;
  float lc = 0.f;
  float* p = fg + bh * 1024;
  for (int s = 0; s < 1024; ++s) { lc += logsigf(p[s]); p[s] = lc; }
}

// Flash-style mLSTM attention. One wave per 16-row tile per (b,h).
__global__ void __launch_bounds__(32)
k_attn(const float* __restrict__ q, const float* __restrict__ kk,
       const float* __restrict__ vv, const float* __restrict__ lc,
       const float* __restrict__ ig, float* __restrict__ hc) {
  __shared__ _Float16 ldsVT[64 * 32];  // V chunk transposed: [n(0..63)][k(0..31)]
  __shared__ _Float16 ldsP[16 * 32];   // probability tile
  int lane = threadIdx.x;
  int tile = blockIdx.x & 63;
  int bh = blockIdx.x >> 6;
  int b = bh >> 2, hd = bh & 3;
  int s0 = tile * 16;
  int n = lane & 15, half = lane >> 4;

  const float* qbase = q + (size_t)(b * 1024 + s0) * 256 + hd * 64;
  const float* kbase = kk + (size_t)(b * 1024) * 256 + hd * 64;
  const float* vbase = vv + (size_t)(b * 1024) * 256 + hd * 64;
  const float* lcp = lc + bh * 1024;
  const float* igp = ig + bh * 1024;

  v16h aq0 = load_frag_f32(qbase, 256, lane);
  v16h aq1 = load_frag_f32(qbase + 32, 256, lane);

  float lcs[8], mrow[8], srow[8], alpha[8];
  v8f acc[4];
  #pragma unroll
  for (int r = 0; r < 8; ++r) {
    lcs[r] = lcp[s0 + r + 8 * half];
    mrow[r] = -1e30f; srow[r] = 0.f;
  }
  #pragma unroll
  for (int nt = 0; nt < 4; ++nt) acc[nt] = (v8f){};

  int tend = s0 + 16;
  for (int t0 = 0; t0 < tend; t0 += 32) {
    // prefetch next chunk's K and V rows (global_prefetch_b8; speculative OK)
    if (t0 + 32 < tend) {
      __builtin_prefetch(kbase + (size_t)(t0 + 32 + lane) * 256, 0, 3);
      __builtin_prefetch(vbase + (size_t)(t0 + 32 + lane) * 256, 0, 3);
    }

    // qk tiles (16 rows x 32 t-cols), K=64 chained WMMAs
    v8f c0 = {}, c1 = {};
    c0 = wmma_f16(aq0, load_frag_f32(kbase + (size_t)t0 * 256, 256, lane), c0);
    c0 = wmma_f16(aq1, load_frag_f32(kbase + (size_t)t0 * 256 + 32, 256, lane), c0);
    c1 = wmma_f16(aq0, load_frag_f32(kbase + (size_t)(t0 + 16) * 256, 256, lane), c1);
    c1 = wmma_f16(aq1, load_frag_f32(kbase + (size_t)(t0 + 16) * 256 + 32, 256, lane), c1);

    float lct0 = lcp[t0 + n],      igt0 = igp[t0 + n];
    float lct1 = lcp[t0 + 16 + n], igt1 = igp[t0 + 16 + n];

    // stage V chunk (32 x 64) into LDS transposed as f16
    const float* vrow = vbase + (size_t)(t0 + lane) * 256;
    #pragma unroll
    for (int c4 = 0; c4 < 16; ++c4) {
      float4 vd = *(const float4*)(vrow + c4 * 4);
      ldsVT[(c4 * 4 + 0) * 32 + lane] = (_Float16)vd.x;
      ldsVT[(c4 * 4 + 1) * 32 + lane] = (_Float16)vd.y;
      ldsVT[(c4 * 4 + 2) * 32 + lane] = (_Float16)vd.z;
      ldsVT[(c4 * 4 + 3) * 32 + lane] = (_Float16)vd.w;
    }

    // gating + online max/sum (flash) in f32
    #pragma unroll
    for (int r = 0; r < 8; ++r) {
      int rowi = s0 + r + 8 * half;
      float qk0 = c0[r] * 0.125f, qk1 = c1[r] * 0.125f;
      float l0 = (t0 + n      <= rowi) ? (lcs[r] - lct0 + igt0) : -1e30f;
      float l1 = (t0 + 16 + n <= rowi) ? (lcs[r] - lct1 + igt1) : -1e30f;
      float cm = rmax16(fmaxf(l0, l1));
      float mnew = fmaxf(mrow[r], cm);
      float al = __expf(mrow[r] - mnew);
      float p0 = qk0 * __expf(l0 - mnew);
      float p1 = qk1 * __expf(l1 - mnew);
      srow[r] = srow[r] * al + rsum16(p0 + p1);
      mrow[r] = mnew; alpha[r] = al;
      ldsP[(r + 8 * half) * 32 + n]      = (_Float16)p0;
      ldsP[(r + 8 * half) * 32 + 16 + n] = (_Float16)p1;
    }

    // fence LDS stores (P + V^T) before fragment gathers; single-wave WG so
    // the barrier degenerates to the dscnt wait we need.
    __syncthreads();

    // out = out*alpha + P(16x32) @ V(32x64)
    v16h aP = load_frag_f16(ldsP, 32, lane);
    #pragma unroll
    for (int nt = 0; nt < 4; ++nt) {
      v16h bv = load_frag_f16(ldsVT + nt * 16 * 32, 32, lane);
      v8f t = acc[nt];
      #pragma unroll
      for (int r = 0; r < 8; ++r) t[r] *= alpha[r];
      acc[nt] = wmma_f16(aP, bv, t);
    }

    // fence fragment gathers before the next chunk overwrites LDS
    __syncthreads();
  }

  #pragma unroll
  for (int r = 0; r < 8; ++r) {
    float denom = fmaxf(fabsf(srow[r]), __expf(-mrow[r])) + 1e-6f;
    size_t orow = (size_t)(b * 1024 + s0 + r + 8 * half) * 256 + hd * 64;
    #pragma unroll
    for (int nt = 0; nt < 4; ++nt)
      hc[orow + nt * 16 + n] = acc[nt][r] / denom;
  }
}

// hs = (mh_ln(hc, onw, groups of 64) + skip*xc) * silu(z)
__global__ void k_mhln_gate(const float* __restrict__ hc, const float* __restrict__ onw,
                            const float* __restrict__ skip, const float* __restrict__ xc,
                            const float* __restrict__ up, float* __restrict__ hs) {
  __shared__ float red[256];
  int row = blockIdx.x, tid = threadIdx.x;
  float xv = hc[(size_t)row * 256 + tid];
  red[tid] = xv; __syncthreads();
  for (int s = 32; s > 0; s >>= 1) { if ((tid & 63) < s) red[tid] += red[tid + s]; __syncthreads(); }
  float mu = red[tid & ~63] * (1.f / 64.f); __syncthreads();
  float d = xv - mu;
  red[tid] = d * d; __syncthreads();
  for (int s = 32; s > 0; s >>= 1) { if ((tid & 63) < s) red[tid] += red[tid + s]; __syncthreads(); }
  float var = red[tid & ~63] * (1.f / 64.f);
  float ht = d * rsqrtf(var + 1e-5f) * onw[tid];
  float z = up[(size_t)row * 512 + 256 + tid];
  hs[(size_t)row * 256 + tid] = (ht + skip[tid] * xc[(size_t)row * 256 + tid]) * siluf(z);
}

// sLSTM sequential scan: one block per (b,h); thread (g,e); state in LDS/regs.
__global__ void __launch_bounds__(128)
k_slstm(const float* __restrict__ gi, const float* __restrict__ gf,
        const float* __restrict__ gz, const float* __restrict__ go,
        const float* __restrict__ R, const float* __restrict__ bb,
        float* __restrict__ ys) {
  __shared__ float yl[32];
  __shared__ float rawl[128];
  int b = blockIdx.x >> 2, hh = blockIdx.x & 3;
  int tid = threadIdx.x;
  int g = tid >> 5, e = tid & 31;
  float Rr[32];
  #pragma unroll
  for (int d = 0; d < 32; ++d)
    Rr[d] = R[((size_t)(hh * 32 + d) * 4 + g) * 32 + e];
  float bg = bb[(hh * 4 + g) * 32 + e];
  const float* gp = (g == 0) ? gi : (g == 1) ? gf : (g == 2) ? gz : go;
  if (tid < 32) yl[tid] = 0.f;
  float cst = 0.f, nst = 0.f, mst = 0.f;
  __syncthreads();
  for (int t = 0; t < 1024; ++t) {
    size_t base = (size_t)(b * 1024 + t) * 128 + hh * 32 + e;
    float acc = gp[base] + bg;
    #pragma unroll
    for (int d = 0; d < 32; ++d) acc += yl[d] * Rr[d];
    rawl[tid] = acc;
    __syncthreads();
    if (tid < 32) {
      float ir = rawl[tid], fr = rawl[32 + tid], zr = rawl[64 + tid], orr = rawl[96 + tid];
      float lfm = mst + logsigf(fr);
      float mnew = (t == 0) ? ir : fmaxf(ir, lfm);
      float og = sigmoidf_(orr);
      float igv = __expf(ir - mnew);
      float fgv = __expf(lfm - mnew);
      cst = fgv * cst + igv * tanhf(zr);
      nst = fgv * nst + igv;
      mst = mnew;
      float yn = og * cst / nst;
      yl[tid] = yn;
      ys[(size_t)(b * 1024 + t) * 128 + hh * 32 + tid] = yn;
    }
    __syncthreads();
  }
}

// h += mh_ln(ys, gnw, groups of 32). One wave == one group.
__global__ void k_mhln_res(const float* __restrict__ ys, const float* __restrict__ gnw,
                           float* __restrict__ h) {
  int row = blockIdx.x, tid = threadIdx.x;
  float xv = ys[(size_t)row * 128 + tid];
  float s = xv;
  #pragma unroll
  for (int o = 1; o < 32; o <<= 1) s += __shfl_xor(s, o, 32);
  float mu = s * (1.f / 32.f);
  float d = xv - mu;
  float sq = d * d;
  #pragma unroll
  for (int o = 1; o < 32; o <<= 1) sq += __shfl_xor(sq, o, 32);
  float var = sq * (1.f / 32.f);
  h[(size_t)row * 128 + tid] += d * rsqrtf(var + 1e-5f) * gnw[tid];
}

// gu = gelu(u[:, :192]) * u[:, 192:]
__global__ void k_geglu(const float* __restrict__ u, float* __restrict__ gu) {
  int i = blockIdx.x * blockDim.x + threadIdx.x;
  if (i >= 8192 * 192) return;
  int row = i / 192, j = i - row * 192;
  float gv = u[(size_t)row * 384 + j];
  float u2 = u[(size_t)row * 384 + 192 + j];
  gu[(size_t)row * 192 + j] = geluf(gv) * u2;
}

// final: LN(last row, post_w) @ fc_W.T + fc_b
__global__ void k_final(const float* __restrict__ h, const float* __restrict__ pw,
                        const float* __restrict__ fcW, const float* __restrict__ fcb,
                        float* __restrict__ out) {
  __shared__ float red[128];
  int b = blockIdx.x, tid = threadIdx.x;
  size_t row = (size_t)(b * 1024 + 1023) * 128;
  float xv = h[row + tid];
  red[tid] = xv; __syncthreads();
  for (int s = 64; s > 0; s >>= 1) { if (tid < s) red[tid] += red[tid + s]; __syncthreads(); }
  float mu = red[0] * (1.f / 128.f); __syncthreads();
  float d = xv - mu;
  red[tid] = d * d; __syncthreads();
  for (int s = 64; s > 0; s >>= 1) { if (tid < s) red[tid] += red[tid + s]; __syncthreads(); }
  float var = red[0] * (1.f / 128.f); __syncthreads();
  red[tid] = d * rsqrtf(var + 1e-5f) * pw[tid] * fcW[tid];
  __syncthreads();
  for (int s = 64; s > 0; s >>= 1) { if (tid < s) red[tid] += red[tid + s]; __syncthreads(); }
  if (tid == 0) out[b] = red[0] + fcb[0];
}

// ---------------------------------------------------------------- launcher
extern "C" void kernel_launch(void* const* d_in, const int* in_sizes, int n_in,
                              void* d_out, int out_size, void* d_ws, size_t ws_size,
                              hipStream_t stream) {
  const float* x       = (const float*)d_in[0];
  const float* W_in    = (const float*)d_in[1];
  const float* b_in    = (const float*)d_in[2];
  const float* ln0_w   = (const float*)d_in[3];
  const float* m_Wup   = (const float*)d_in[4];
  const float* m_bup   = (const float*)d_in[5];
  const float* m_convk = (const float*)d_in[6];
  const float* m_convb = (const float*)d_in[7];
  const float* m_Wq    = (const float*)d_in[8];
  const float* m_bq    = (const float*)d_in[9];
  const float* m_Wk    = (const float*)d_in[10];
  const float* m_bk    = (const float*)d_in[11];
  const float* m_Wv    = (const float*)d_in[12];
  const float* m_bv    = (const float*)d_in[13];
  const float* m_Wig   = (const float*)d_in[14];
  const float* m_big   = (const float*)d_in[15];
  const float* m_Wfg   = (const float*)d_in[16];
  const float* m_bfg   = (const float*)d_in[17];
  const float* m_onw   = (const float*)d_in[18];
  const float* m_skip  = (const float*)d_in[19];
  const float* m_Wd    = (const float*)d_in[20];
  const float* m_bd    = (const float*)d_in[21];
  const float* ln1_w   = (const float*)d_in[22];
  const float* s_convk = (const float*)d_in[23];
  const float* s_convb = (const float*)d_in[24];
  const float* s_Wi    = (const float*)d_in[25];
  const float* s_Wf    = (const float*)d_in[26];
  const float* s_Wz    = (const float*)d_in[27];
  const float* s_Wo    = (const float*)d_in[28];
  const float* s_R     = (const float*)d_in[29];
  const float* s_b     = (const float*)d_in[30];
  const float* s_gnw   = (const float*)d_in[31];
  const float* ffn_nw  = (const float*)d_in[32];
  const float* f_Wu    = (const float*)d_in[33];
  const float* f_bu    = (const float*)d_in[34];
  const float* f_Wd    = (const float*)d_in[35];
  const float* f_bd    = (const float*)d_in[36];
  const float* post_w  = (const float*)d_in[37];
  const float* fc_W    = (const float*)d_in[38];
  const float* fc_b    = (const float*)d_in[39];
  float* out = (float*)d_out;
  float* ws = (float*)d_ws;

  // workspace layout (floats); aliased regions noted
  const size_t OFF_H  = 0;                    // 1M
  const size_t OFF_R  = 1048576;              // 1M (later aliased by ig/fg)
  const size_t OFF_UP = 2097152;              // 4M (later aliased by ys/rc)
  const size_t OFF_XC = 6291456;              // 2M
  const size_t OFF_Q  = 8388608;              // 2M (later aliased by u)
  const size_t OFF_K  = 10485760;             // 2M
  const size_t OFF_V  = 12582912;             // 2M (later aliased by gu)
  const size_t OFF_HC = 14680064;             // 2M (later aliased by gi/gf)
  const size_t OFF_HS = 16777216;             // 2M (later aliased by gz/go)
  float* h  = ws + OFF_H;
  float* r  = ws + OFF_R;
  float* up = ws + OFF_UP;
  float* xc = ws + OFF_XC;
  float* q  = ws + OFF_Q;
  float* k  = ws + OFF_K;
  float* v  = ws + OFF_V;
  float* hc = ws + OFF_HC;
  float* hs = ws + OFF_HS;
  float* igb = ws + OFF_R;            // 32K, r is dead in this window
  float* fgb = ws + OFF_R + 32768;    // 32K, becomes lc after cumsum
  float* ysb = ws + OFF_UP;           // 1M, up dead by sLSTM phase
  float* rc  = ws + OFF_UP + 1048576; // 1M
  float* gib = ws + OFF_HC;           // 1M
  float* gfb = ws + OFF_HC + 1048576; // 1M
  float* gzb = ws + OFF_HS;           // 1M
  float* gob = ws + OFF_HS + 1048576; // 1M
  float* u   = ws + OFF_Q;            // 3M, q+k dead by FFN phase
  float* gu  = ws + OFF_V;            // 1.5M, v dead by FFN phase

  // ---- embed + LN + up-proj ----
  k_embed<<<8192, 128, 0, stream>>>(x, W_in, b_in, h);
  k_ln128<<<8192, 128, 0, stream>>>(h, ln0_w, r);
  k_gemm16<false><<<dim3(512, 32), 32, 0, stream>>>(r, 128, m_Wup, m_bup, nullptr, up, 512, 128);

  // ---- mLSTM front-end ----
  k_conv<<<8192, 256, 0, stream>>>(up, 0, 512, m_convk, m_convb, xc, 256);
  k_headwise<<<8192, 256, 0, stream>>>(xc, 0, 256, m_Wq, m_bq, q, 64, 4, 4);
  k_headwise<<<8192, 256, 0, stream>>>(xc, 0, 256, m_Wk, m_bk, k, 64, 4, 4);
  k_headwise<<<8192, 256, 0, stream>>>(up, 0, 512, m_Wv, m_bv, v, 64, 4, 4);
  k_gates<<<256, 256, 0, stream>>>(q, k, v, m_Wig, m_big, m_Wfg, m_bfg, igb, fgb);
  k_fgcum<<<1, 32, 0, stream>>>(fgb);

  // ---- mLSTM flash attention (WMMA) ----
  k_attn<<<2048, 32, 0, stream>>>(q, k, v, fgb, igb, hc);

  // ---- gate + down-proj (+residual) ----
  k_mhln_gate<<<8192, 256, 0, stream>>>(hc, m_onw, m_skip, xc, up, hs);
  k_gemm16<true><<<dim3(512, 8), 32, 0, stream>>>(hs, 256, m_Wd, m_bd, h, h, 128, 256);

  // ---- sLSTM ----
  k_ln128<<<8192, 128, 0, stream>>>(h, ln1_w, r);
  k_conv<<<4096, 256, 0, stream>>>(r, 0, 128, s_convk, s_convb, rc, 128);
  k_headwise<<<4096, 256, 0, stream>>>(rc, 0, 128, s_Wi, nullptr, gib, 4, 32, 32);
  k_headwise<<<4096, 256, 0, stream>>>(rc, 0, 128, s_Wf, nullptr, gfb, 4, 32, 32);
  k_headwise<<<4096, 256, 0, stream>>>(r, 0, 128, s_Wz, nullptr, gzb, 4, 32, 32);
  k_headwise<<<4096, 256, 0, stream>>>(r, 0, 128, s_Wo, nullptr, gob, 4, 32, 32);
  k_slstm<<<32, 128, 0, stream>>>(gib, gfb, gzb, gob, s_R, s_b, ysb);
  k_mhln_res<<<8192, 128, 0, stream>>>(ysb, s_gnw, h);

  // ---- FFN ----
  k_ln128<<<8192, 128, 0, stream>>>(h, ffn_nw, r);
  k_gemm16<false><<<dim3(512, 24), 32, 0, stream>>>(r, 128, f_Wu, f_bu, nullptr, u, 384, 128);
  k_geglu<<<6144, 256, 0, stream>>>(u, gu);
  k_gemm16<true><<<dim3(512, 8), 32, 0, stream>>>(gu, 192, f_Wd, f_bd, h, h, 128, 192);

  // ---- head ----
  k_final<<<8, 128, 0, stream>>>(h, post_w, fc_W, fc_b, out);

  (void)in_sizes; (void)n_in; (void)out_size; (void)ws_size;
}
